// BertWithRopeAttention_14173392077010
// MI455X (gfx1250) — compile-verified, matrix operand order checked by reading, not run on hardware
//
#include <hip/hip_runtime.h>
#include <hip/hip_bf16.h>

#define B_   4
#define S_   2048
#define H_   1024
#define HH_  16
#define DH_  64

typedef __attribute__((ext_vector_type(16))) __bf16 v16bf;
typedef __attribute__((ext_vector_type(8)))  float  v8f;

union FragBF {
    v16bf v;
    uint4 u[2];
};

__device__ __forceinline__ unsigned short f2bf(float f) {
    unsigned u = __float_as_uint(f);
    u += 0x7FFFu + ((u >> 16) & 1u);   // round-to-nearest-even
    return (unsigned short)(u >> 16);
}
__device__ __forceinline__ float bf2f(unsigned short s) {
    return __uint_as_float(((unsigned)s) << 16);
}

__device__ __forceinline__ void store_out(unsigned short* p, float v) { *p = f2bf(v); }
__device__ __forceinline__ void store_out(float* p, float v) { *p = v; }

// Async DMA: global -> LDS, 16 bytes per lane. Tracked by ASYNCcnt.
__device__ __forceinline__ void async_ld_b128(unsigned lds_off, const void* gaddr) {
    asm volatile("global_load_async_to_lds_b128 %0, %1, off"
                 :: "v"(lds_off), "v"(gaddr) : "memory");
}
__device__ __forceinline__ void wait_async0() {
    asm volatile("s_wait_asynccnt 0x0" ::: "memory");
}
__device__ __forceinline__ unsigned lds_off_of(const void* p) {
    return (unsigned)(unsigned long long)p;   // low 32 bits = LDS byte offset
}

// ---------------------------------------------------------------------------
// f32 -> bf16 (vectorized x4)
// ---------------------------------------------------------------------------
__global__ void f32_to_bf16_kernel(const float* __restrict__ src,
                                   unsigned short* __restrict__ dst, int n4) {
    int i = blockIdx.x * blockDim.x + threadIdx.x;
    if (i >= n4) return;
    float4 f = ((const float4*)src)[i];
    ushort4 o;
    o.x = f2bf(f.x); o.y = f2bf(f.y); o.z = f2bf(f.z); o.w = f2bf(f.w);
    ((ushort4*)dst)[i] = o;
}

// ---------------------------------------------------------------------------
// transpose KxN f32 -> NxK bf16   (so GEMM B-operand is N-major "Bt")
// ---------------------------------------------------------------------------
__global__ void transpose_to_bf16_kernel(const float* __restrict__ src,
                                         unsigned short* __restrict__ dst,
                                         int K, int N) {
    long i = (long)blockIdx.x * blockDim.x + threadIdx.x;
    long total = (long)K * N;
    if (i >= total) return;
    int k = (int)(i / N);
    int n = (int)(i % N);
    dst[(long)n * K + k] = f2bf(src[i]);
}

// ---------------------------------------------------------------------------
// bf16 GEMM: C[M,N] = A[M,K] * Bt[N,K]^T + bias, fp32 accumulate via WMMA.
// 256 threads (8 waves), tile 128x128, K-step 32, double-buffered LDS with
// register staging: global loads for step kt+32 issue before the WMMA block.
// ---------------------------------------------------------------------------
template <typename OutT>
__global__ __launch_bounds__(256) void gemm_bf16_kernel(
    const unsigned short* __restrict__ A,
    const unsigned short* __restrict__ Bt,
    const float* __restrict__ bias,
    OutT* __restrict__ C,
    int M, int N, int K) {
    const int LDSS = 48;  // row stride in halves: 96B, multiple of 16B
    __shared__ unsigned short As[2][128 * LDSS];
    __shared__ unsigned short Bs[2][128 * LDSS];

    const int mBase = blockIdx.y * 128;
    const int nBase = blockIdx.x * 128;
    const int t = threadIdx.x;
    const int w = t >> 5, lane = t & 31, g = lane >> 4, ln = lane & 15;
    const int wr = w >> 1, wc = w & 1;

    v8f acc[2][4];
    const v8f vzero = {0.f, 0.f, 0.f, 0.f, 0.f, 0.f, 0.f, 0.f};
#pragma unroll
    for (int mi = 0; mi < 2; mi++)
#pragma unroll
        for (int ni = 0; ni < 4; ni++) acc[mi][ni] = vzero;

    uint4 ra[2], rb[2];
    const int row0 = t >> 2, cs0 = (t & 3) * 8;
    const int row1 = (t + 256) >> 2, cs1 = ((t + 256) & 3) * 8;

    ra[0] = *(const uint4*)(A + (long)(mBase + row0) * K + cs0);
    rb[0] = *(const uint4*)(Bt + (long)(nBase + row0) * K + cs0);
    ra[1] = *(const uint4*)(A + (long)(mBase + row1) * K + cs1);
    rb[1] = *(const uint4*)(Bt + (long)(nBase + row1) * K + cs1);
    *(uint4*)(&As[0][row0 * LDSS + cs0]) = ra[0];
    *(uint4*)(&Bs[0][row0 * LDSS + cs0]) = rb[0];
    *(uint4*)(&As[0][row1 * LDSS + cs1]) = ra[1];
    *(uint4*)(&Bs[0][row1 * LDSS + cs1]) = rb[1];
    __syncthreads();

    int buf = 0;
    for (int kt = 0; kt < K; kt += 32) {
        const bool hasNext = (kt + 32 < K);
        if (hasNext) {  // issue next tile's global loads (overlap with WMMA)
            ra[0] = *(const uint4*)(A + (long)(mBase + row0) * K + kt + 32 + cs0);
            rb[0] = *(const uint4*)(Bt + (long)(nBase + row0) * K + kt + 32 + cs0);
            ra[1] = *(const uint4*)(A + (long)(mBase + row1) * K + kt + 32 + cs1);
            rb[1] = *(const uint4*)(Bt + (long)(nBase + row1) * K + kt + 32 + cs1);
        }

        FragBF afr[2], bfr[4];
#pragma unroll
        for (int mi = 0; mi < 2; mi++) {
            int r = wr * 32 + mi * 16 + ln;
            afr[mi].u[0] = *(const uint4*)(&As[buf][r * LDSS + 8 * g]);
            afr[mi].u[1] = *(const uint4*)(&As[buf][r * LDSS + 16 + 8 * g]);
        }
#pragma unroll
        for (int ni = 0; ni < 4; ni++) {
            int r = wc * 64 + ni * 16 + ln;
            bfr[ni].u[0] = *(const uint4*)(&Bs[buf][r * LDSS + 8 * g]);
            bfr[ni].u[1] = *(const uint4*)(&Bs[buf][r * LDSS + 16 + 8 * g]);
        }
#pragma unroll
        for (int mi = 0; mi < 2; mi++)
#pragma unroll
            for (int ni = 0; ni < 4; ni++)
                acc[mi][ni] = __builtin_amdgcn_wmma_f32_16x16x32_bf16(
                    false, afr[mi].v, false, bfr[ni].v, (short)0, acc[mi][ni],
                    false, false);

        if (hasNext) {
            *(uint4*)(&As[buf ^ 1][row0 * LDSS + cs0]) = ra[0];
            *(uint4*)(&Bs[buf ^ 1][row0 * LDSS + cs0]) = rb[0];
            *(uint4*)(&As[buf ^ 1][row1 * LDSS + cs1]) = ra[1];
            *(uint4*)(&Bs[buf ^ 1][row1 * LDSS + cs1]) = rb[1];
        }
        __syncthreads();
        buf ^= 1;
    }

#pragma unroll
    for (int mi = 0; mi < 2; mi++) {
#pragma unroll
        for (int ni = 0; ni < 4; ni++) {
            int col = nBase + wc * 64 + ni * 16 + ln;
            float bv = bias ? bias[col] : 0.f;
#pragma unroll
            for (int r = 0; r < 8; r++) {
                int row = mBase + wr * 32 + mi * 16 + r + 8 * g;
                store_out(&C[(long)row * N + col], acc[mi][ni][r] + bv);
            }
        }
    }
}

// ---------------------------------------------------------------------------
// RoPE + split + layout: qkv[M,3072] bf16 -> Q[b,h,s,d] (pre-scaled by
// 1/sqrt(Dh)), K[b,h,s,d], Vt[b,h,d,s]  (all bf16).
// ---------------------------------------------------------------------------
__global__ void rope_split_kernel(const int* __restrict__ positions,
                                  const unsigned short* __restrict__ qkv,
                                  unsigned short* __restrict__ Q,
                                  unsigned short* __restrict__ Kd,
                                  unsigned short* __restrict__ Vt) {
    int idx = blockIdx.x * blockDim.x + threadIdx.x;  // < 1<<22
    int j = idx & 31;
    int h = (idx >> 5) & 15;
    int s = (idx >> 9) & (S_ - 1);
    int b = idx >> 20;

    long row = (long)b * S_ + s;
    const unsigned short* rp = qkv + row * 3072;
    int off = h * 64 + j;
    float q1 = bf2f(rp[off]),         q2 = bf2f(rp[off + 32]);
    float k1 = bf2f(rp[1024 + off]),  k2 = bf2f(rp[1024 + off + 32]);
    float v1 = bf2f(rp[2048 + off]),  v2 = bf2f(rp[2048 + off + 32]);

    float inv = __expf(-(float)j * 0.2878231366242557f);  // ln(1e4)/32
    float ang = (float)positions[s] * inv;
    float sn, cs;
    __sincosf(ang, &sn, &cs);

    const float scale = 0.125f;  // Dh^-0.5 folded into Q
    long bh = (long)b * HH_ + h;
    long qb = (bh * S_ + s) * 64;
    Q[qb + j]       = f2bf((q1 * cs - q2 * sn) * scale);
    Q[qb + 32 + j]  = f2bf((q2 * cs + q1 * sn) * scale);
    Kd[qb + j]      = f2bf(k1 * cs - k2 * sn);
    Kd[qb + 32 + j] = f2bf(k2 * cs + k1 * sn);
    Vt[(bh * 64 + j) * S_ + s]      = f2bf(v1);
    Vt[(bh * 64 + j + 32) * S_ + s] = f2bf(v2);
}

// ---------------------------------------------------------------------------
// Flash attention. Grid (B*HH, S/128), 8 waves/block, wave w owns query rows
// [w*16, w*16+16). K/V 64-key tiles staged in LDS shared by all 8 waves,
// double-buffered, filled with global_load_async_to_lds_b128 prefetch.
// Row sums computed on the matrix pipe (P x ones WMMA) instead of bpermute.
// ---------------------------------------------------------------------------
#define AL 80  // halves per K/V row in LDS: 160B, multiple of 16B

__global__ __launch_bounds__(256) void attn_kernel(
    const unsigned short* __restrict__ Q,
    const unsigned short* __restrict__ Kd,
    const unsigned short* __restrict__ Vt,
    unsigned short* __restrict__ ctx) {
    __shared__ unsigned short Ks[2][64 * AL];   // [key][d]
    __shared__ unsigned short Vs[2][64 * AL];   // [d][key]
    __shared__ unsigned short Plds[8][16 * 64]; // per-wave P tile, XOR-swizzled

    const int bh = blockIdx.x;  // 0..63
    const int b = bh >> 4, h = bh & 15;
    const int t = threadIdx.x;
    const int w = t >> 5, lane = t & 31;
    const int g = lane >> 4, ln = lane & 15;
    const int qTile = blockIdx.y * 128 + w * 16;

    const unsigned short* Qp = Q + ((long)bh * S_ + qTile) * 64;
    const unsigned short* Kp = Kd + (long)bh * S_ * 64;
    const unsigned short* Vp = Vt + (long)bh * 64 * S_;
    unsigned short* Pw = &Plds[w][0];

    const int srow0 = t >> 3,         scs0 = (t & 7) * 8;
    const int srow1 = (t + 256) >> 3, scs1 = ((t + 256) & 7) * 8;

    // Q fragments (d-chunks [0,32) and [32,64)) straight from global.
    FragBF qf[2];
#pragma unroll
    for (int dc = 0; dc < 2; dc++) {
        const unsigned short* p = Qp + (long)ln * 64 + dc * 32;
        qf[dc].u[0] = *(const uint4*)(p + 8 * g);
        qf[dc].u[1] = *(const uint4*)(p + 16 + 8 * g);
    }

    // all-ones bf16 B-matrix fragment: row-sum via WMMA (D = P x 1)
    FragBF ones;
    ones.u[0] = make_uint4(0x3F803F80u, 0x3F803F80u, 0x3F803F80u, 0x3F803F80u);
    ones.u[1] = ones.u[0];

    float mrow[8], alpha[8];
    v8f O[4], lacc;
    const v8f vzero = {0.f, 0.f, 0.f, 0.f, 0.f, 0.f, 0.f, 0.f};
#pragma unroll
    for (int r = 0; r < 8; r++) mrow[r] = -1e30f;
#pragma unroll
    for (int d = 0; d < 4; d++) O[d] = vzero;
    lacc = vzero;

    // prologue: DMA chunk 0 into buffer 0
    async_ld_b128(lds_off_of(&Ks[0][srow0 * AL + scs0]), Kp + (long)srow0 * 64 + scs0);
    async_ld_b128(lds_off_of(&Vs[0][srow0 * AL + scs0]), Vp + (long)srow0 * S_ + scs0);
    async_ld_b128(lds_off_of(&Ks[0][srow1 * AL + scs1]), Kp + (long)srow1 * 64 + scs1);
    async_ld_b128(lds_off_of(&Vs[0][srow1 * AL + scs1]), Vp + (long)srow1 * S_ + scs1);
    wait_async0();
    __syncthreads();

    int buf = 0;
    for (int kc = 0; kc < S_; kc += 64) {
        // ---- prefetch next 64-key chunk into the other buffer ----
        if (kc + 64 < S_) {
            const unsigned short* kn = Kp + (long)(kc + 64) * 64;
            const unsigned short* vn = Vp + kc + 64;
            async_ld_b128(lds_off_of(&Ks[buf ^ 1][srow0 * AL + scs0]), kn + (long)srow0 * 64 + scs0);
            async_ld_b128(lds_off_of(&Vs[buf ^ 1][srow0 * AL + scs0]), vn + (long)srow0 * S_ + scs0);
            async_ld_b128(lds_off_of(&Ks[buf ^ 1][srow1 * AL + scs1]), kn + (long)srow1 * 64 + scs1);
            async_ld_b128(lds_off_of(&Vs[buf ^ 1][srow1 * AL + scs1]), vn + (long)srow1 * S_ + scs1);
        }

        // ---- scores S = Q K^T (scale folded into Q) ----
        v8f sc[4];
#pragma unroll
        for (int jj = 0; jj < 4; jj++) {
            sc[jj] = vzero;
            const unsigned short* kp = &Ks[buf][(jj * 16 + ln) * AL];
            FragBF kf0, kf1;
            kf0.u[0] = *(const uint4*)(kp + 8 * g);
            kf0.u[1] = *(const uint4*)(kp + 16 + 8 * g);
            kf1.u[0] = *(const uint4*)(kp + 32 + 8 * g);
            kf1.u[1] = *(const uint4*)(kp + 48 + 8 * g);
            sc[jj] = __builtin_amdgcn_wmma_f32_16x16x32_bf16(
                false, qf[0].v, false, kf0.v, (short)0, sc[jj], false, false);
            sc[jj] = __builtin_amdgcn_wmma_f32_16x16x32_bf16(
                false, qf[1].v, false, kf1.v, (short)0, sc[jj], false, false);
        }

        // ---- online max + exp (row m = r + 8g lives across 16 lanes) ----
#pragma unroll
        for (int r = 0; r < 8; r++) {
            float mx = fmaxf(fmaxf(sc[0][r], sc[1][r]), fmaxf(sc[2][r], sc[3][r]));
#pragma unroll
            for (int msk = 8; msk >= 1; msk >>= 1) mx = fmaxf(mx, __shfl_xor(mx, msk));
            float mnew = fmaxf(mrow[r], mx);
            alpha[r] = __expf(mrow[r] - mnew);
            mrow[r] = mnew;
#pragma unroll
            for (int jj = 0; jj < 4; jj++) sc[jj][r] = __expf(sc[jj][r] - mnew);
        }
#pragma unroll
        for (int r = 0; r < 8; r++) {
            lacc[r] *= alpha[r];
#pragma unroll
            for (int d = 0; d < 4; d++) O[d][r] *= alpha[r];
        }

        // ---- C-layout -> A-layout shuffle of P via per-wave LDS ----
#pragma unroll
        for (int jj = 0; jj < 4; jj++) {
            int col = jj * 16 + ln;
            int chunk = col >> 3, within = col & 7;
#pragma unroll
            for (int r = 0; r < 8; r++) {
                int m = r + 8 * g;
                int pc = chunk ^ (m & 7);
                Pw[m * 64 + pc * 8 + within] = f2bf(sc[jj][r]);
            }
        }
        asm volatile("s_wait_dscnt 0x0" ::: "memory");  // wave-local LDS RAW

        FragBF pa[2];
#pragma unroll
        for (int kk = 0; kk < 2; kk++) {
            int c0 = kk * 4 + g;
            int c1 = kk * 4 + 2 + g;
            pa[kk].u[0] = *(const uint4*)(Pw + ln * 64 + ((c0 ^ (ln & 7)) * 8));
            pa[kk].u[1] = *(const uint4*)(Pw + ln * 64 + ((c1 ^ (ln & 7)) * 8));
        }

        // ---- row sums on the matrix pipe: lacc += P x ones ----
        lacc = __builtin_amdgcn_wmma_f32_16x16x32_bf16(
            false, pa[0].v, false, ones.v, (short)0, lacc, false, false);
        lacc = __builtin_amdgcn_wmma_f32_16x16x32_bf16(
            false, pa[1].v, false, ones.v, (short)0, lacc, false, false);

        // ---- ctx += P * V ----
#pragma unroll
        for (int j2 = 0; j2 < 4; j2++) {
            const unsigned short* vp = &Vs[buf][(j2 * 16 + ln) * AL];
            FragBF vf0, vf1;
            vf0.u[0] = *(const uint4*)(vp + 8 * g);
            vf0.u[1] = *(const uint4*)(vp + 16 + 8 * g);
            vf1.u[0] = *(const uint4*)(vp + 32 + 8 * g);
            vf1.u[1] = *(const uint4*)(vp + 48 + 8 * g);
            O[j2] = __builtin_amdgcn_wmma_f32_16x16x32_bf16(
                false, pa[0].v, false, vf0.v, (short)0, O[j2], false, false);
            O[j2] = __builtin_amdgcn_wmma_f32_16x16x32_bf16(
                false, pa[1].v, false, vf1.v, (short)0, O[j2], false, false);
        }

        wait_async0();     // prefetch landed
        __syncthreads();   // everyone done with current buffer
        buf ^= 1;
    }

    // ---- finalize: multiply by 1/rowsum, write ctx[b,s, h*64+d] ----
    float rinv[8];
#pragma unroll
    for (int r = 0; r < 8; r++) rinv[r] = 1.0f / lacc[r];
#pragma unroll
    for (int j2 = 0; j2 < 4; j2++) {
        int col = h * 64 + j2 * 16 + ln;
#pragma unroll
        for (int r = 0; r < 8; r++) {
            int srow = qTile + r + 8 * g;
            ctx[((long)b * S_ + srow) * H_ + col] = f2bf(O[j2][r] * rinv[r]);
        }
    }
}

// ---------------------------------------------------------------------------
// Host launcher
// ---------------------------------------------------------------------------
extern "C" void kernel_launch(void* const* d_in, const int* in_sizes, int n_in,
                              void* d_out, int out_size, void* d_ws, size_t ws_size,
                              hipStream_t stream) {
    (void)in_sizes; (void)n_in; (void)out_size; (void)ws_size;
    const int*   positions = (const int*)d_in[0];
    const float* hidden    = (const float*)d_in[1];
    const float* Wqkv      = (const float*)d_in[2];
    const float* bqkv      = (const float*)d_in[3];
    const float* Wout      = (const float*)d_in[4];
    const float* bout      = (const float*)d_in[5];
    float* out = (float*)d_out;

    char* ws = (char*)d_ws;
    unsigned short* Xbf   = (unsigned short*)(ws);                // 16,777,216
    unsigned short* WqkvT = (unsigned short*)(ws + 16777216);     //  6,291,456
    unsigned short* WoutT = (unsigned short*)(ws + 23068672);     //  2,097,152
    unsigned short* qkv   = (unsigned short*)(ws + 25165824);     // 50,331,648
    unsigned short* Qb    = (unsigned short*)(ws + 75497472);     // 16,777,216
    unsigned short* Kb    = (unsigned short*)(ws + 92274688);     // 16,777,216
    unsigned short* Vtb   = (unsigned short*)(ws + 109051904);    // 16,777,216
    unsigned short* ctx   = (unsigned short*)(ws + 125829120);    // 16,777,216

    f32_to_bf16_kernel<<<8192, 256, 0, stream>>>(hidden, Xbf, 2097152);
    transpose_to_bf16_kernel<<<12288, 256, 0, stream>>>(Wqkv, WqkvT, 1024, 3072);
    transpose_to_bf16_kernel<<<4096, 256, 0, stream>>>(Wout, WoutT, 1024, 1024);

    gemm_bf16_kernel<unsigned short><<<dim3(24, 64), 256, 0, stream>>>(
        Xbf, WqkvT, bqkv, qkv, 8192, 3072, 1024);

    rope_split_kernel<<<16384, 256, 0, stream>>>(positions, qkv, Qb, Kb, Vtb);

    attn_kernel<<<dim3(64, 16), 256, 0, stream>>>(Qb, Kb, Vtb, ctx);

    gemm_bf16_kernel<float><<<dim3(8, 64), 256, 0, stream>>>(
        ctx, WoutT, bout, out, 8192, 1024, 1024);
}
